// PiNet2_61993557950617
// MI455X (gfx1250) — matrix operand order repaired and checked by reading, not compiled, not run on hardware
//
#include <hip/hip_runtime.h>
#include <hip/hip_bf16.h>
#include <stdint.h>

// ---------------------------------------------------------------------------
// Problem constants
// ---------------------------------------------------------------------------
constexpr int   NA      = 50000;
constexpr int   NP      = 800000;
constexpr int   NA_PAD  = 50048;           // 782 blocks * 4 waves * 16 atoms
constexpr float RC_F    = 4.0f;
constexpr float PI_F    = 3.14159265358979323846f;

// ---------------------------------------------------------------------------
// WMMA types / helpers (gfx1250: v_wmma_f32_16x16x32_f16)
// ---------------------------------------------------------------------------
typedef __attribute__((ext_vector_type(16))) _Float16 v16h;
typedef __attribute__((ext_vector_type(8)))  float    v8f;

union V16H { v16h v; uint4 q[2]; };
union F8   { v8f  v; float f[8]; };

#define WB() __builtin_amdgcn_wave_barrier()

// Branch-free tanh on the hardware transcendental pipe:
// tanh(x) = (e-1)/(e+1), e = exp2(2*log2(e)*x).  Clamp keeps exp2 finite;
// |x|>9 is already +-1 to fp32 precision.
__device__ __forceinline__ float fast_tanh(float x) {
  x = fminf(9.0f, fmaxf(-9.0f, x));
  float e = __builtin_amdgcn_exp2f(x * 2.8853900817779268f);  // 2/ln(2)
  return (e - 1.0f) * __builtin_amdgcn_rcpf(e + 1.0f);
}

// A fragment (16x32 f16, M=rows): lane<16 -> K 0-7 & 16-23, lane>=16 -> K 8-15 & 24-31
__device__ __forceinline__ v16h ld_afrag(const _Float16* tile, int lane) {
  const int row = lane & 15;
  const int hi  = (lane >> 4) & 1;
  V16H r;
  r.q[0] = *reinterpret_cast<const uint4*>(tile + row * 32 + hi * 8);
  r.q[1] = *reinterpret_cast<const uint4*>(tile + row * 32 + 16 + hi * 8);
  return r.v;
}

// B fragment (32x16 f16, stored transposed [N][K=32]): lane carries column N=lane&15,
// lane<16 -> K 0-7 & 16-23, lane>=16 -> K 8-15 & 24-31.
__device__ __forceinline__ v16h ld_bfrag(const _Float16* __restrict__ wT, int lane) {
  const int n  = lane & 15;
  const int hi = (lane >> 4) & 1;
  V16H r;
  r.q[0] = *reinterpret_cast<const uint4*>(wT + n * 32 + hi * 8);
  r.q[1] = *reinterpret_cast<const uint4*>(wT + n * 32 + 16 + hi * 8);
  return r.v;
}

__device__ __forceinline__ v8f wmma16(v16h a, v16h b, v8f c) {
  return __builtin_amdgcn_wmma_f32_16x16x32_f16(
      /*neg_a=*/false, a, /*neg_b=*/false, b,
      /*c_mod=*/(short)0, c, /*reuse_a=*/false, /*reuse_b=*/false);
}

// ---------------------------------------------------------------------------
// Workspace layout
// ---------------------------------------------------------------------------
// f16 transposed-weight block per layer (offsets in halves)
constexpr int OFF_piW1T  = 0;     // [16][32]
constexpr int OFF_piW2T  = 512;   // [64][32]
constexpr int OFF_iiW1T  = 2560;  // [16][32]
constexpr int OFF_iiW2T  = 3072;  // [32][32]
constexpr int OFF_pixT   = 4096;  // [16][32]  (K: 0-15 = wi rows, 16-31 = wj rows)
constexpr int OFF_ppW1T  = 4608;  // [16][32]
constexpr int OFF_ppW2T  = 5120;  // [16][32]
constexpr int OFF_eqT    = 5632;  // [16][32]
constexpr int OFF_dotT   = 6144;  // [16][32]
constexpr int OFF_gcW1T  = 6656;  // [16][32]
constexpr int OFF_gcW2T  = 7168;  // [32][32]
constexpr int OFF_outW1T = 8192;  // [16][32]
constexpr int OFF_outW2T = 8704;  // [16][32]
constexpr int WLAYER_H   = 9216;

// fp32 bias/small block per layer (offsets in floats)
constexpr int B_pi_b1 = 0, B_pi_b2 = 16, B_gc_b1 = 80, B_gc_b2 = 96;
constexpr int B_out_b1 = 128, B_out_b2 = 144, B_out_W = 160, B_out_b = 176;
constexpr int B_res1 = 192;      // d0 only, 4x16
constexpr int BLAYER_F = 256;

constexpr size_t WS_W_OFF = 0;
constexpr size_t WS_B_OFF = WS_W_OFF + (size_t)4 * WLAYER_H * 2;
constexpr size_t WS_P1F   = WS_B_OFF + (size_t)4 * BLAYER_F * 4;
constexpr size_t WS_P1H   = WS_P1F + (size_t)NA_PAD * 16 * 4;
constexpr size_t WS_P3F   = WS_P1H + (size_t)NA_PAD * 16 * 2;
constexpr size_t WS_P3H   = WS_P3F + (size_t)NA_PAD * 48 * 4;
constexpr size_t WS_P1N   = WS_P3H + (size_t)NA_PAD * 48 * 2;
constexpr size_t WS_P3N   = WS_P1N + (size_t)NA_PAD * 32 * 4;

// ---------------------------------------------------------------------------
// Parameter pointer table (pytree leaves, dict keys sorted)
// ---------------------------------------------------------------------------
struct LayerP {
  const float *dot_W, *eq_pp_W, *gc_W1, *gc_W2, *gc_b1, *gc_b2, *ii_W1, *ii_W2,
              *out_W, *out_W1, *out_W2, *out_b, *out_b1, *out_b2,
              *pi_W1, *pi_W2, *pi_b1, *pi_b2, *pix_wi, *pix_wj, *pp_W1, *pp_W2,
              *res1_W, *res3_W;
};
struct AllP { LayerP L[4]; };

// ---------------------------------------------------------------------------
// Utility kernels
// ---------------------------------------------------------------------------
__global__ void fill_zero(float* __restrict__ p, size_t n) {
  size_t i = (size_t)blockIdx.x * blockDim.x + threadIdx.x;
  size_t stride = (size_t)gridDim.x * blockDim.x;
  for (size_t t = i; t < n; t += stride) p[t] = 0.0f;
}

// Convert fp32 weights to f16, transposed [N][K=32] with zero padding.
__global__ void conv_w(AllP P, _Float16* __restrict__ wh, float* __restrict__ bf) {
  const int d = blockIdx.x;
  const LayerP L = P.L[d];
  _Float16* W = wh + (size_t)d * WLAYER_H;
  float* Bf = bf + (size_t)d * BLAYER_F;
  const int t = threadIdx.x;
  const int cin = (d == 0) ? 4 : 16;
  const int c3  = (d == 0) ? 1 : 16;

  auto putT = [&](int off, const float* src, int K, int N) {
    for (int idx = t; idx < N * 32; idx += 256) {
      int n = idx >> 5, k = idx & 31;
      W[off + n * 32 + k] = (_Float16)((k < K) ? src[k * N + n] : 0.0f);
    }
  };

  // pi_W1: concat(p1i,p1j) laid out as K 0..15 = i-feats, 16..31 = j-feats
  for (int idx = t; idx < 16 * 32; idx += 256) {
    int n = idx >> 5, k = idx & 31;
    float v = 0.0f;
    if (k < cin)                        v = L.pi_W1[k * 16 + n];
    else if (k >= 16 && (k - 16) < cin) v = L.pi_W1[(cin + k - 16) * 16 + n];
    W[OFF_piW1T + n * 32 + k] = (_Float16)v;
  }
  putT(OFF_piW2T, L.pi_W2, 16, 64);
  putT(OFF_iiW1T, L.ii_W1, 16, 16);
  putT(OFF_iiW2T, L.ii_W2, 16, 32);
  // pix: K 0..c3-1 = wi rows, 16..16+c3-1 = wj rows
  for (int idx = t; idx < 16 * 32; idx += 256) {
    int n = idx >> 5, k = idx & 31;
    float v = 0.0f;
    if (n < c3) {
      if (k < c3)                        v = L.pix_wi[k * c3 + n];
      else if (k >= 16 && (k - 16) < c3) v = L.pix_wj[(k - 16) * c3 + n];
    }
    W[OFF_pixT + n * 32 + k] = (_Float16)v;
  }
  putT(OFF_ppW1T,  L.pp_W1,   32, 16);
  putT(OFF_ppW2T,  L.pp_W2,   16, 16);
  putT(OFF_eqT,    L.eq_pp_W, 16, 16);
  putT(OFF_dotT,   L.dot_W,   16, 16);
  putT(OFF_gcW1T,  L.gc_W1,   32, 16);
  putT(OFF_gcW2T,  L.gc_W2,   16, 32);
  putT(OFF_outW1T, L.out_W1,  16, 16);
  putT(OFF_outW2T, L.out_W2,  16, 16);

  for (int i = t; i < 16; i += 256) Bf[B_pi_b1 + i] = L.pi_b1[i];
  for (int i = t; i < 64; i += 256) Bf[B_pi_b2 + i] = L.pi_b2[i];
  for (int i = t; i < 16; i += 256) Bf[B_gc_b1 + i] = L.gc_b1[i];
  for (int i = t; i < 32; i += 256) Bf[B_gc_b2 + i] = L.gc_b2[i];
  for (int i = t; i < 16; i += 256) Bf[B_out_b1 + i] = L.out_b1[i];
  for (int i = t; i < 16; i += 256) Bf[B_out_b2 + i] = L.out_b2[i];
  for (int i = t; i < 16; i += 256) Bf[B_out_W + i] = L.out_W[i];
  if (t == 0) Bf[B_out_b] = L.out_b[0];
  if (d == 0) for (int i = t; i < 64; i += 256) Bf[B_res1 + i] = L.res1_W[i];
}

// Per-layer prep: (d==0) materialize padded p1 fp32; always refresh f16 mirrors.
__global__ void prep_kernel(int d, const float* __restrict__ p1in,
                            float* __restrict__ p1f, _Float16* __restrict__ p1h,
                            const float* __restrict__ p3f, _Float16* __restrict__ p3h) {
  size_t i = (size_t)blockIdx.x * blockDim.x + threadIdx.x;
  size_t stride = (size_t)gridDim.x * blockDim.x;
  const size_t n1 = (size_t)NA_PAD * 16;
  const size_t n3 = (size_t)NA_PAD * 48;
  for (size_t t = i; t < n1; t += stride) {
    float v;
    if (d == 0) {
      size_t a = t >> 4; int c = (int)(t & 15);
      v = (a < (size_t)NA && c < 4) ? p1in[a * 4 + c] : 0.0f;
      p1f[t] = v;
    } else {
      v = p1f[t];
    }
    p1h[t] = (_Float16)v;
  }
  for (size_t t = i; t < n3; t += stride) p3h[t] = (_Float16)p3f[t];
}

// ---------------------------------------------------------------------------
// Pair kernel: one wave = 16 pairs.  11 WMMAs per wave.
// ---------------------------------------------------------------------------
__global__ __launch_bounds__(128) void pair_kernel(
    const _Float16* __restrict__ p1h, const _Float16* __restrict__ p3h,
    const float* __restrict__ dist, const float* __restrict__ diff,
    const int* __restrict__ ind2,
    const _Float16* __restrict__ W, const float* __restrict__ B,
    float* __restrict__ p1n, float* __restrict__ p3n) {
  __shared__ __align__(16) _Float16 sA32[4][16][32];
  __shared__ __align__(16) _Float16 sA16[4][16][32];
  __shared__ __align__(16) float    sT64[4][16][64];
  __shared__ __align__(16) float    sT2 [4][16][16];
  __shared__ int   sI[4][16];
  __shared__ int   sJ[4][16];
  __shared__ float sBAS[4][16][4];
  __shared__ float sD3 [4][16][3];

  const int w    = threadIdx.x >> 5;
  const int lane = threadIdx.x & 31;
  const int row  = lane & 15;
  const int hi   = lane >> 4;
  const int pairBase = (blockIdx.x * 4 + w) * 16;

  _Float16 (*A32)[32] = sA32[w];
  _Float16 (*A16)[32] = sA16[w];
  float (*T64)[64] = sT64[w];
  float (*T2)[16]  = sT2[w];
  int* II = sI[w]; int* JJ = sJ[w];
  float (*BAS)[4] = sBAS[w];
  float (*D3)[3]  = sD3[w];

  if (lane < 16) {
    int p = pairBase + lane;
    II[lane] = ind2[2 * p];
    JJ[lane] = ind2[2 * p + 1];
    float dd = dist[p];
    // fc = 0.5*(cos(pi*d/4)+1) = 0.5*(cos(2*pi*(d/8))+1): native v_cos_f32
    float fc = 0.5f * (__builtin_amdgcn_cosf(dd * 0.125f) + 1.0f);
    float f2 = fc * fc;
    BAS[lane][0] = fc; BAS[lane][1] = f2; BAS[lane][2] = f2 * fc; BAS[lane][3] = f2 * f2;
    float dx = diff[3 * p], dy = diff[3 * p + 1], dz = diff[3 * p + 2];
    float inv = __builtin_amdgcn_rsqf(dx * dx + dy * dy + dz * dz);
    D3[lane][0] = dx * inv; D3[lane][1] = dy * inv; D3[lane][2] = dz * inv;
    uint4 z = {0, 0, 0, 0};                       // permanent zero pad: K 16..31
    *reinterpret_cast<uint4*>(&A16[lane][16]) = z;
    *reinterpret_cast<uint4*>(&A16[lane][24]) = z;
  }
  WB();

  const v8f zeroc = {0, 0, 0, 0, 0, 0, 0, 0};

  // --- gather pair features [p1(i) | p1(j)] into 16x32 f16 A-tile
  {
    int a = hi ? JJ[row] : II[row];
    const uint4* src = reinterpret_cast<const uint4*>(p1h + (size_t)a * 16);
    uint4 q0 = src[0], q1 = src[1];
    uint4* dst = reinterpret_cast<uint4*>(&A32[row][hi * 16]);
    dst[0] = q0; dst[1] = q1;
  }
  WB();

  // --- pi layer 1: (2cin->16) tanh
  {
    v16h a = ld_afrag(&A32[0][0], lane);
    v16h b = ld_bfrag(W + OFF_piW1T, lane);
    F8 acc; acc.v = wmma16(a, b, zeroc);
    float bias = B[B_pi_b1 + row];
#pragma unroll
    for (int v = 0; v < 8; v++)
      A16[v + 8 * hi][row] = (_Float16)fast_tanh(acc.f[v] + bias);
  }
  WB();

  // --- pi layer 2: (16->64) tanh -> T64
  {
    v16h a = ld_afrag(&A16[0][0], lane);
#pragma unroll
    for (int g = 0; g < 4; g++) {
      v16h b = ld_bfrag(W + OFF_piW2T + g * 512, lane);
      F8 acc; acc.v = wmma16(a, b, zeroc);
      float bias = B[B_pi_b2 + g * 16 + row];
#pragma unroll
      for (int v = 0; v < 8; v++)
        T64[v + 8 * hi][g * 16 + row] = fast_tanh(acc.f[v] + bias);
    }
  }
  WB();

  // --- basis contraction: inter(P,16,4) . basis(P,4) -> A16 cols 0..15
  {
    float b0 = BAS[row][0], b1 = BAS[row][1], b2 = BAS[row][2], b3 = BAS[row][3];
#pragma unroll
    for (int cc = 0; cc < 8; cc++) {
      int c = 8 * hi + cc;
      const float* tp = &T64[row][4 * c];
      A16[row][c] = (_Float16)(tp[0] * b0 + tp[1] * b1 + tp[2] * b2 + tp[3] * b3);
    }
  }
  WB();

  // --- ii_W1: (16->16) tanh
  {
    v16h a = ld_afrag(&A16[0][0], lane);
    v16h b = ld_bfrag(W + OFF_iiW1T, lane);
    F8 acc; acc.v = wmma16(a, b, zeroc);
#pragma unroll
    for (int v = 0; v < 8; v++)
      A16[v + 8 * hi][row] = (_Float16)fast_tanh(acc.f[v]);
  }
  WB();

  // --- ii_W2: (16->32) tanh -> atomic segment-sum p1n; keep i1b = i1[16:32]
  {
    v16h a = ld_afrag(&A16[0][0], lane);
#pragma unroll
    for (int g = 0; g < 2; g++) {
      v16h b = ld_bfrag(W + OFF_iiW2T + g * 512, lane);
      F8 acc; acc.v = wmma16(a, b, zeroc);
#pragma unroll
      for (int v = 0; v < 8; v++) {
        int m = v + 8 * hi;
        float val = fast_tanh(acc.f[v]);
        atomicAdd(&p1n[(size_t)II[m] * 32 + g * 16 + row], val);
        if (g == 1) T2[m][row] = val;
      }
    }
  }
  WB();

  // --- equivariant channel: ix = (p3i@wi + p3j@wj + d3) * i1b, scatter into p3n
  const v16h bpix = ld_bfrag(W + OFF_pixT, lane);   // loop-invariant B fragment
  for (int x = 0; x < 3; x++) {
    {
      int a = hi ? JJ[row] : II[row];
      const uint4* src = reinterpret_cast<const uint4*>(p3h + ((size_t)a * 3 + x) * 16);
      uint4 q0 = src[0], q1 = src[1];
      uint4* dst = reinterpret_cast<uint4*>(&A32[row][hi * 16]);
      dst[0] = q0; dst[1] = q1;
    }
    WB();
    v16h a = ld_afrag(&A32[0][0], lane);
    F8 acc; acc.v = wmma16(a, bpix, zeroc);
#pragma unroll
    for (int v = 0; v < 8; v++) {
      int m = v + 8 * hi;
      float val = (acc.f[v] + D3[m][x]) * T2[m][row];
      atomicAdd(&p3n[((size_t)II[m] * 3 + x) * 16 + row], val);
    }
    WB();
  }
}

// ---------------------------------------------------------------------------
// Atom kernel: one wave = 16 atoms.  13 WMMAs per wave.
// ---------------------------------------------------------------------------
__global__ __launch_bounds__(128) void atom_kernel(
    int d, const float* __restrict__ p1in,
    const _Float16* __restrict__ W, const float* __restrict__ B,
    const float* __restrict__ p1n, const float* __restrict__ p3n,
    float* __restrict__ p1f, float* __restrict__ p3f,
    float* __restrict__ dout) {
  __shared__ __align__(16) _Float16 sA32[4][16][32];
  __shared__ __align__(16) _Float16 sA16[4][16][32];
  __shared__ __align__(16) float    sP3 [4][3][16][16];

  const int w    = threadIdx.x >> 5;
  const int lane = threadIdx.x & 31;
  const int row  = lane & 15;
  const int hi   = lane >> 4;
  const int aBase = (blockIdx.x * 4 + w) * 16;

  _Float16 (*A32)[32] = sA32[w];
  _Float16 (*A16)[32] = sA16[w];
  float (*P3)[16][16] = sP3[w];

  if (lane < 16) {
    uint4 z = {0, 0, 0, 0};
    *reinterpret_cast<uint4*>(&A16[lane][16]) = z;
    *reinterpret_cast<uint4*>(&A16[lane][24]) = z;
  }

  // p1n (32 fp32) -> A32 f16
  {
    const float* src = p1n + (size_t)(aBase + row) * 32 + hi * 16;
#pragma unroll
    for (int k = 0; k < 16; k++) A32[row][hi * 16 + k] = (_Float16)src[k];
  }
  WB();

  const v8f zeroc = {0, 0, 0, 0, 0, 0, 0, 0};

  // pp_W1 (32->16) tanh
  {
    v16h a = ld_afrag(&A32[0][0], lane);
    v16h b = ld_bfrag(W + OFF_ppW1T, lane);
    F8 acc; acc.v = wmma16(a, b, zeroc);
#pragma unroll
    for (int v = 0; v < 8; v++)
      A16[v + 8 * hi][row] = (_Float16)fast_tanh(acc.f[v]);
  }
  WB();
  // pp_W2 (16->16) tanh -> A32 cols 0..15 (gc input lower half)
  {
    v16h a = ld_afrag(&A16[0][0], lane);
    v16h b = ld_bfrag(W + OFF_ppW2T, lane);
    F8 acc; acc.v = wmma16(a, b, zeroc);
#pragma unroll
    for (int v = 0; v < 8; v++)
      A32[v + 8 * hi][row] = (_Float16)fast_tanh(acc.f[v]);
  }
  WB();

  // equivariant: p3n' = p3n @ eq_pp_W; dotted = sum_x (p3n'@dot_W) * p3n'
  float dotacc[8] = {0, 0, 0, 0, 0, 0, 0, 0};
  const v16h beq  = ld_bfrag(W + OFF_eqT, lane);    // loop-invariant B fragments
  const v16h bdot = ld_bfrag(W + OFF_dotT, lane);
  for (int x = 0; x < 3; x++) {
    if (lane < 16) {
      const float* src = p3n + ((size_t)(aBase + lane) * 3 + x) * 16;
#pragma unroll
      for (int k = 0; k < 16; k++) A16[lane][k] = (_Float16)src[k];
    }
    WB();
    {
      v16h a = ld_afrag(&A16[0][0], lane);
      F8 acc; acc.v = wmma16(a, beq, zeroc);
#pragma unroll
      for (int v = 0; v < 8; v++) {
        int m = v + 8 * hi;
        P3[x][m][row] = acc.f[v];
        A16[m][row] = (_Float16)acc.f[v];
      }
    }
    WB();
    {
      v16h a = ld_afrag(&A16[0][0], lane);
      F8 q; q.v = wmma16(a, bdot, zeroc);
#pragma unroll
      for (int v = 0; v < 8; v++) dotacc[v] += q.f[v] * P3[x][v + 8 * hi][row];
    }
    WB();
  }
  // dotted -> A32 cols 16..31
#pragma unroll
  for (int v = 0; v < 8; v++) A32[v + 8 * hi][16 + row] = (_Float16)dotacc[v];
  WB();

  // gc1 (32->16) tanh
  {
    v16h a = ld_afrag(&A32[0][0], lane);
    v16h b = ld_bfrag(W + OFF_gcW1T, lane);
    F8 acc; acc.v = wmma16(a, b, zeroc);
    float bias = B[B_gc_b1 + row];
#pragma unroll
    for (int v = 0; v < 8; v++)
      A16[v + 8 * hi][row] = (_Float16)fast_tanh(acc.f[v] + bias);
  }
  WB();
  // gc2 (16->32) tanh -> p1t, s3
  float p1t[8], s3[8];
  {
    v16h a = ld_afrag(&A16[0][0], lane);
    v16h b0 = ld_bfrag(W + OFF_gcW2T, lane);
    F8 a0; a0.v = wmma16(a, b0, zeroc);
    float bias0 = B[B_gc_b2 + row];
#pragma unroll
    for (int v = 0; v < 8; v++) p1t[v] = fast_tanh(a0.f[v] + bias0);
    v16h b1 = ld_bfrag(W + OFF_gcW2T + 512, lane);
    F8 a1; a1.v = wmma16(a, b1, zeroc);
    float bias1 = B[B_gc_b2 + 16 + row];
#pragma unroll
    for (int v = 0; v < 8; v++) s3[v] = fast_tanh(a1.f[v] + bias1);
  }

  // p3 <- p3 + p3n' * s3   (d==0: p3 buffer is zero, res3 term vanishes since p3==0)
#pragma unroll
  for (int x = 0; x < 3; x++) {
#pragma unroll
    for (int v = 0; v < 8; v++) {
      int m = v + 8 * hi;
      size_t a = (size_t)(aBase + m);
      float* dst = &p3f[(a * 3 + x) * 16 + row];
      *dst = *dst + P3[x][m][row] * s3[v];
    }
  }
  // p1 <- (d==0 ? p1_in @ res1_W : p1) + p1t
#pragma unroll
  for (int v = 0; v < 8; v++) {
    int m = v + 8 * hi;
    int a = aBase + m;
    float base = 0.0f;
    if (d == 0) {
      if (a < NA) {
        const float* r1 = B + B_res1;
        const float* pin = p1in + (size_t)a * 4;
#pragma unroll
        for (int r = 0; r < 4; r++) base += pin[r] * r1[r * 16 + row];
      }
    } else {
      base = p1f[(size_t)a * 16 + row];
    }
    p1f[(size_t)a * 16 + row] = base + p1t[v];
  }

  // out chain: tanh(p1t@W1+b1) -> tanh(@W2+b2) -> @out_W + out_b, atomic into dout
  WB();
#pragma unroll
  for (int v = 0; v < 8; v++) A16[v + 8 * hi][row] = (_Float16)p1t[v];
  WB();
  {
    v16h a = ld_afrag(&A16[0][0], lane);
    v16h b = ld_bfrag(W + OFF_outW1T, lane);
    F8 acc; acc.v = wmma16(a, b, zeroc);
    float bias = B[B_out_b1 + row];
#pragma unroll
    for (int v = 0; v < 8; v++)
      A16[v + 8 * hi][row] = (_Float16)fast_tanh(acc.f[v] + bias);
  }
  WB();
  {
    v16h a = ld_afrag(&A16[0][0], lane);
    v16h b = ld_bfrag(W + OFF_outW2T, lane);
    F8 acc; acc.v = wmma16(a, b, zeroc);
    float bias = B[B_out_b2 + row];
    float wout = B[B_out_W + row];
    float ob   = B[B_out_b];
#pragma unroll
    for (int v = 0; v < 8; v++) {
      int a2 = aBase + v + 8 * hi;
      if (a2 < NA) {
        float o = fast_tanh(acc.f[v] + bias);
        float contrib = o * wout + ((row == 0) ? ob : 0.0f);
        atomicAdd(&dout[a2], contrib);
      }
    }
  }
}

// ---------------------------------------------------------------------------
// Host driver
// ---------------------------------------------------------------------------
extern "C" void kernel_launch(void* const* d_in, const int* in_sizes, int n_in,
                              void* d_out, int out_size, void* d_ws, size_t ws_size,
                              hipStream_t stream) {
  (void)in_sizes; (void)n_in; (void)out_size; (void)ws_size;

  const float* p1   = (const float*)d_in[0];
  const float* dist = (const float*)d_in[1];
  const float* diff = (const float*)d_in[2];
  const int*   ind2 = (const int*)d_in[94];   // ind_1 at 93 is unused

  AllP P{};
  const int base[4] = {3, 27, 49, 71};        // layer pytree-leaf bases
  for (int dd = 0; dd < 4; dd++) {
    auto g = [&](int k) { return (const float*)d_in[base[dd] + k]; };
    LayerP& L = P.L[dd];
    L.dot_W = g(0);  L.eq_pp_W = g(1); L.gc_W1 = g(2);  L.gc_W2 = g(3);
    L.gc_b1 = g(4);  L.gc_b2 = g(5);   L.ii_W1 = g(6);  L.ii_W2 = g(7);
    L.out_W = g(8);  L.out_W1 = g(9);  L.out_W2 = g(10); L.out_b = g(11);
    L.out_b1 = g(12); L.out_b2 = g(13); L.pi_W1 = g(14); L.pi_W2 = g(15);
    L.pi_b1 = g(16); L.pi_b2 = g(17);  L.pix_wi = g(18); L.pix_wj = g(19);
    L.pp_W1 = g(20); L.pp_W2 = g(21);
    if (dd == 0) { L.res1_W = g(22); L.res3_W = g(23); }
    else         { L.res1_W = nullptr; L.res3_W = nullptr; }
  }

  uint8_t* ws = (uint8_t*)d_ws;
  _Float16* wh  = (_Float16*)(ws + WS_W_OFF);
  float*    bf  = (float*)(ws + WS_B_OFF);
  float*    p1f = (float*)(ws + WS_P1F);
  _Float16* p1h = (_Float16*)(ws + WS_P1H);
  float*    p3f = (float*)(ws + WS_P3F);
  _Float16* p3h = (_Float16*)(ws + WS_P3H);
  float*    p1n = (float*)(ws + WS_P1N);
  float*    p3n = (float*)(ws + WS_P3N);
  float*    dout = (float*)d_out;

  fill_zero<<<256, 256, 0, stream>>>(dout, (size_t)NA);
  fill_zero<<<1024, 256, 0, stream>>>(p3f, (size_t)NA_PAD * 48);
  conv_w<<<4, 256, 0, stream>>>(P, wh, bf);

  for (int dd = 0; dd < 4; dd++) {
    _Float16* Wd = wh + (size_t)dd * WLAYER_H;
    float*    Bd = bf + (size_t)dd * BLAYER_F;
    fill_zero<<<1024, 256, 0, stream>>>(p1n, (size_t)NA_PAD * 32);
    fill_zero<<<1024, 256, 0, stream>>>(p3n, (size_t)NA_PAD * 48);
    prep_kernel<<<2048, 256, 0, stream>>>(dd, p1, p1f, p1h, p3f, p3h);
    pair_kernel<<<NP / 64, 128, 0, stream>>>(p1h, p3h, dist, diff, ind2, Wd, Bd, p1n, p3n);
    atom_kernel<<<NA_PAD / 64, 128, 0, stream>>>(dd, p1, Wd, Bd, p1n, p3n, p1f, p3f, dout);
  }
}
